// WorkingMemoryNet_83176336654730
// MI455X (gfx1250) — matrix operand order repaired and checked by reading, not compiled, not run on hardware
//
#include <hip/hip_runtime.h>

// GRU on MI455X (gfx1250): f16-in / f32-acc WMMA for all three GEMMs.
// B=512, T=512, I=H=O=256.

typedef __attribute__((ext_vector_type(16))) _Float16 v16h;
typedef __attribute__((ext_vector_type(8)))  float    v8f;
typedef __attribute__((ext_vector_type(4)))  _Float16 v4h;
typedef __attribute__((ext_vector_type(2)))  _Float16 v2h;

#define LDS_STRIDE 264   // 16x256 f16 tile rows padded: 264*2B=528B -> distinct banks per row

__device__ __forceinline__ float fast_sigmoid(float x) {
    return 1.0f / (1.0f + __expf(-x));
}
__device__ __forceinline__ float fast_tanh(float x) {
    return 1.0f - 2.0f / (__expf(2.0f * x) + 1.0f);   // saturates correctly at +/-1
}

// A-fragment (16x32 f16, MxK) from LDS per ISA 7.12.2:
//   lane: m = lane&15, half = lane>>4
//   VGPR p<4 : K = half*8 + 2p, +1 ; VGPR p>=4: K = 16 + half*8 + 2(p-4), +1
__device__ __forceinline__ v16h load_a_frag(const _Float16* lds, int lane, int kbase) {
    const int m = lane & 15;
    const int half = lane >> 4;
    v16h a;
#pragma unroll
    for (int p = 0; p < 8; ++p) {
        const int k = kbase + ((p < 4) ? (half * 8 + 2 * p)
                                       : (16 + half * 8 + 2 * (p - 4)));
        v2h t = *(const v2h*)(lds + m * LDS_STRIDE + k);
        a[2 * p]     = t[0];
        a[2 * p + 1] = t[1];
    }
    return a;
}

// B-fragment from pre-packed panel: one contiguous 32B load per lane.
// Panel layout (pack_w): ((tj*8 + kk)*32 + lane)*16 halfs.  K assumed 256 (KT=8).
__device__ __forceinline__ v16h load_b_frag(const _Float16* __restrict__ panel,
                                            int tj, int kk, int lane) {
    return *(const v16h*)(panel + (((size_t)(tj * 8 + kk) * 32 + lane) << 4));
}

__device__ __forceinline__ v8f wmma_f16(v16h a, v16h b, v8f c) {
    return __builtin_amdgcn_wmma_f32_16x16x32_f16(false, a, false, b,
                                                  (short)0, c, false, false);
}

// ---- Pack W (G rows x K cols, row-major f32) into WMMA B-panel f16 layout. ----
// B-fragment element i (0..15) of lane l maps to K = kk*32 + (i<8 ? (l>>4)*8 + i : 16 + (l>>4)*8 + i-8),
// output col (W row) = tj*16 + (l&15).
__global__ void gru_pack_w(const float* __restrict__ W, _Float16* __restrict__ panel,
                           int G, int K) {
    int idx = blockIdx.x * blockDim.x + threadIdx.x;
    int total = G * K;
    if (idx >= total) return;
    int i    = idx & 15;
    int l    = (idx >> 4) & 31;
    int rest = idx >> 9;
    int KT   = K >> 5;
    int kk   = rest % KT;
    int tj   = rest / KT;
    int k    = kk * 32 + ((i < 8) ? ((l >> 4) * 8 + i) : (16 + (l >> 4) * 8 + (i - 8)));
    int g    = tj * 16 + (l & 15);
    panel[idx] = (_Float16)W[(size_t)g * K + k];
}

__global__ void gru_zero_f32(float* __restrict__ p, int n) {
    int i = blockIdx.x * blockDim.x + threadIdx.x;
    if (i < n) p[i] = 0.0f;
}

// ---- One GRU timestep.  grid=32 (batch tiles), block=512 (16 waves). ----
// Wave j owns hidden tile j (16 cols); computes gx_{r,z,n} and gh_{r,z,n}
// (6 f32 accumulators), then the gate nonlinearity, updating hbuf in place
// and writing hiddens[:, t, :].
__global__ __launch_bounds__(512) void gru_step_kernel(
    const float* __restrict__ inputs,         // (B,T,I) f32
    const _Float16* __restrict__ Wih_p,       // packed (768x256)
    const _Float16* __restrict__ Whh_p,       // packed (768x256)
    const float* __restrict__ b_ih,           // (768)
    const float* __restrict__ b_hh,           // (768)
    float* __restrict__ hbuf,                 // (512,256) f32, updated in place
    float* __restrict__ hiddens,              // (B,T,H) f32 output
    int t) {
    __shared__ _Float16 xs[16 * LDS_STRIDE];
    __shared__ _Float16 hs[16 * LDS_STRIDE];

    const int tid   = threadIdx.x;
    const int lane  = tid & 31;
    const int wave  = tid >> 5;          // hidden tile j = 0..15
    const int bbase = blockIdx.x * 16;   // batch rows [bbase, bbase+16)

    // Stage x_t and h into LDS as f16 (each thread: 8 contiguous elements).
    {
        const int row = tid >> 5;
        const int col = (tid & 31) * 8;
        const float* xp = inputs + (((size_t)(bbase + row) * 512 + t) << 8) + col;
        const float* hp = hbuf + ((size_t)(bbase + row) << 8) + col;
#pragma unroll
        for (int q = 0; q < 8; q += 4) {
            float4 xv = *(const float4*)(xp + q);
            float4 hv = *(const float4*)(hp + q);
            v4h xo, ho;
            xo[0] = (_Float16)xv.x; xo[1] = (_Float16)xv.y;
            xo[2] = (_Float16)xv.z; xo[3] = (_Float16)xv.w;
            ho[0] = (_Float16)hv.x; ho[1] = (_Float16)hv.y;
            ho[2] = (_Float16)hv.z; ho[3] = (_Float16)hv.w;
            *(v4h*)(xs + row * LDS_STRIDE + col + q) = xo;
            *(v4h*)(hs + row * LDS_STRIDE + col + q) = ho;
        }
    }
    __syncthreads();

    v8f axr = {}, axz = {}, axn = {};   // x @ W_ih^T  (r,z,n gate tiles)
    v8f ahr = {}, ahz = {}, ahn = {};   // h @ W_hh^T
    const int j = wave;

#pragma unroll
    for (int kk = 0; kk < 8; ++kk) {    // K = 256 = 8 x 32
        v16h ax = load_a_frag(xs, lane, kk * 32);
        v16h ah = load_a_frag(hs, lane, kk * 32);
        v16h br = load_b_frag(Wih_p, 0 * 16 + j, kk, lane);
        v16h bz = load_b_frag(Wih_p, 1 * 16 + j, kk, lane);
        v16h bn = load_b_frag(Wih_p, 2 * 16 + j, kk, lane);
        axr = wmma_f16(ax, br, axr);
        axz = wmma_f16(ax, bz, axz);
        axn = wmma_f16(ax, bn, axn);
        v16h cr = load_b_frag(Whh_p, 0 * 16 + j, kk, lane);
        v16h cz = load_b_frag(Whh_p, 1 * 16 + j, kk, lane);
        v16h cn = load_b_frag(Whh_p, 2 * 16 + j, kk, lane);
        ahr = wmma_f16(ah, cr, ahr);
        ahz = wmma_f16(ah, cz, ahz);
        ahn = wmma_f16(ah, cn, ahn);
    }

    // Gate combine.  C/D layout: lane l -> n = l&15, VGPR v -> m = 8*(l>>4)+v.
    const int ncol = j * 16 + (lane & 15);
    const float bxr = b_ih[ncol],       bhr = b_hh[ncol];
    const float bxz = b_ih[256 + ncol], bhz = b_hh[256 + ncol];
    const float bxn = b_ih[512 + ncol], bhn = b_hh[512 + ncol];
    const int mb = (lane >> 4) * 8;

#pragma unroll
    for (int v = 0; v < 8; ++v) {
        const int m = bbase + mb + v;                   // global batch row
        const size_t hidx = ((size_t)m << 8) + ncol;
        const float hold = hbuf[hidx];
        const float r = fast_sigmoid(axr[v] + bxr + ahr[v] + bhr);
        const float z = fast_sigmoid(axz[v] + bxz + ahz[v] + bhz);
        const float n = fast_tanh((axn[v] + bxn) + r * (ahn[v] + bhn));
        const float hnew = (1.0f - z) * n + z * hold;
        hbuf[hidx] = hnew;                              // column-disjoint per wave
        hiddens[(((size_t)m * 512 + t) << 8) + ncol] = hnew;
    }
}

// ---- Output projection: outputs = hiddens @ W_fc^T + b_fc ----
// grid = (B*T)/16 = 16384, block = 512 (16 waves; wave j -> out-col tile j).
__global__ __launch_bounds__(512) void gru_fc_kernel(
    const float* __restrict__ hiddens,        // (B*T, 256) f32
    const _Float16* __restrict__ Wfc_p,       // packed (256x256)
    const float* __restrict__ b_fc,           // (256)
    float* __restrict__ outputs) {            // (B*T, 256) f32
    __shared__ _Float16 as[16 * LDS_STRIDE];

    const int tid   = threadIdx.x;
    const int lane  = tid & 31;
    const int wave  = tid >> 5;
    const size_t rbase = (size_t)blockIdx.x * 16;

    {
        const int row = tid >> 5;
        const int col = (tid & 31) * 8;
        const float* hp = hiddens + ((rbase + row) << 8) + col;
#pragma unroll
        for (int q = 0; q < 8; q += 4) {
            float4 hv = *(const float4*)(hp + q);
            v4h ho;
            ho[0] = (_Float16)hv.x; ho[1] = (_Float16)hv.y;
            ho[2] = (_Float16)hv.z; ho[3] = (_Float16)hv.w;
            *(v4h*)(as + row * LDS_STRIDE + col + q) = ho;
        }
    }
    __syncthreads();

    v8f acc = {};
#pragma unroll
    for (int kk = 0; kk < 8; ++kk) {
        v16h a = load_a_frag(as, lane, kk * 32);
        v16h b = load_b_frag(Wfc_p, wave, kk, lane);
        acc = wmma_f16(a, b, acc);
    }

    const int ncol = wave * 16 + (lane & 15);
    const float bias = b_fc[ncol];
    const int mb = (lane >> 4) * 8;
#pragma unroll
    for (int v = 0; v < 8; ++v) {
        const size_t m = rbase + mb + v;
        outputs[(m << 8) + ncol] = acc[v] + bias;
    }
}

extern "C" void kernel_launch(void* const* d_in, const int* in_sizes, int n_in,
                              void* d_out, int out_size, void* d_ws, size_t ws_size,
                              hipStream_t stream) {
    (void)in_sizes; (void)n_in; (void)out_size; (void)ws_size;
    constexpr int B = 512, T = 512, H = 256, O = 256;

    const float* inputs = (const float*)d_in[0];
    const float* W_ih   = (const float*)d_in[1];   // (768, 256)
    const float* W_hh   = (const float*)d_in[2];   // (768, 256)
    const float* b_ih   = (const float*)d_in[3];   // (768)
    const float* b_hh   = (const float*)d_in[4];   // (768)
    const float* W_fc   = (const float*)d_in[5];   // (256, 256)
    const float* b_fc   = (const float*)d_in[6];   // (256)

    float* outputs = (float*)d_out;                        // (B,T,O)
    float* hiddens = outputs + (size_t)B * T * O;          // (B,T,H)

    // Workspace: packed f16 weight panels + fp32 hidden state.
    _Float16* Wih_p = (_Float16*)d_ws;                     // 768*256 halfs
    _Float16* Whh_p = Wih_p + 768 * 256;
    _Float16* Wfc_p = Whh_p + 768 * 256;                   // 256*256 halfs
    float*    hbuf  = (float*)(Wfc_p + 256 * 256);         // 512*256 f32 (offset 16B-aligned)

    gru_pack_w<<<(768 * 256 + 255) / 256, 256, 0, stream>>>(W_ih, Wih_p, 768, 256);
    gru_pack_w<<<(768 * 256 + 255) / 256, 256, 0, stream>>>(W_hh, Whh_p, 768, 256);
    gru_pack_w<<<(256 * 256 + 255) / 256, 256, 0, stream>>>(W_fc, Wfc_p, 256, 256);
    gru_zero_f32<<<(B * H + 255) / 256, 256, 0, stream>>>(hbuf, B * H);

    for (int t = 0; t < T; ++t) {
        gru_step_kernel<<<B / 16, 512, 0, stream>>>(inputs, Wih_p, Whh_p,
                                                    b_ih, b_hh, hbuf, hiddens, t);
    }

    gru_fc_kernel<<<(B * T) / 16, 512, 0, stream>>>(hiddens, Wfc_p, b_fc, outputs);
}
